// UAVNetA2CEasy_5789615915394
// MI455X (gfx1250) — compile-verified, matrix operand order checked by reading, not run on hardware
//
#include <hip/hip_runtime.h>

// ---------------- types ----------------
typedef __bf16 bf16;
typedef __attribute__((ext_vector_type(16))) __bf16 v16bf;
typedef __attribute__((ext_vector_type(8)))  __bf16 v8bf;
typedef __attribute__((ext_vector_type(8)))  float   v8f;
typedef __attribute__((ext_vector_type(4)))  unsigned u32x4;
typedef __attribute__((ext_vector_type(8)))  unsigned u32x8;

// ---------------- model constants ----------------
#define C_NUM_P 2048
#define C_NUM_A 1024
#define C_STATE 128
#define C_P_S   256
#define C_IN_P  384
#define C_IN_A  128
#define C_OSQ   9
#define C_XW    3456   // IN_P*OSQ
#define C_DSTAT 2304   // P_S*OSQ
#define C_DOBS  1152   // STATE*OSQ
#define NEG_SLOPE 0.2f

// =====================================================================
// Tensor Data Mover: DMA a [tile_rows x 32] bf16 tile (row-major, stride
// `stride_elems`) from global into LDS at byte address lds_addr.
// D# per CDNA5 ISA. Rows beyond rows_rem zero-fill.
// Must be executed wave-uniformly (EXEC ignored by TDM).
// =====================================================================
__device__ __forceinline__ void tdm_load_tile(unsigned lds_addr, const bf16* gptr,
                                              unsigned rows_rem, unsigned row_len,
                                              unsigned stride_elems, unsigned tile_rows)
{
    unsigned long long ga = (unsigned long long)(const void*)gptr;
    u32x4 g0;
    g0[0] = 1u;                                   // count=1, user mode, no gather
    g0[1] = lds_addr;                             // LDS byte address
    g0[2] = (unsigned)ga;                         // global_addr[31:0]
    g0[3] = (unsigned)((ga >> 32) & 0x1FFFFFFull) // global_addr[56:32]
          | (2u << 30);                           // type = 2 ("image")
    u32x8 g1;
    g1[0] = (1u << 16);                                           // data_size=1 (2B), mask=0
    g1[1] = (row_len & 0xFFFFu) << 16;                            // tensor_dim0[15:0]
    g1[2] = ((row_len >> 16) & 0xFFFFu) | ((rows_rem & 0xFFFFu) << 16); // td0 hi | td1 lo
    g1[3] = ((rows_rem >> 16) & 0xFFFFu) | (32u << 16);           // td1 hi | tile_dim0=32
    g1[4] = tile_rows & 0xFFFFu;                                  // tile_dim1, tile_dim2=0
    g1[5] = stride_elems;                                         // tensor_dim0_stride lo32
    g1[6] = 0u;
    g1[7] = 0u;
    asm volatile("tensor_load_to_lds %0, %1" :: "s"(g0), "s"(g1) : "memory");
}

// =====================================================================
// elementwise helpers
// =====================================================================
__global__ void k_gather(const float* __restrict__ x,
                         bf16* __restrict__ xps,
                         bf16* __restrict__ xpo,
                         bf16* __restrict__ xas)
{
    int tid = blockIdx.x * blockDim.x + threadIdx.x;
    int stride = gridDim.x * blockDim.x;
    for (int i = tid; i < C_NUM_P * C_DSTAT; i += stride) {
        int p = i / C_DSTAT, c = i % C_DSTAT;
        int j = c / C_P_S, k = c % C_P_S;
        xps[i] = (bf16)x[(long)p * C_XW + j * C_IN_P + k];
    }
    for (int i = tid; i < C_NUM_P * C_DOBS; i += stride) {
        int p = i / C_DOBS, c = i % C_DOBS;
        int j = c / C_STATE, k = c % C_STATE;
        xpo[i] = (bf16)x[(long)p * C_XW + (j + 1) * C_P_S + k];
    }
    for (int i = tid; i < C_NUM_A * C_DOBS; i += stride) {
        int a = i / C_DOBS, c = i % C_DOBS;
        int j = c / C_IN_A, k = c % C_IN_A;
        xas[i] = (bf16)x[(long)(C_NUM_P + a) * C_XW + j * C_IN_P + k];
    }
}

__global__ void k_cast(const float* __restrict__ in, bf16* __restrict__ out, long n)
{
    long i = (long)blockIdx.x * blockDim.x + threadIdx.x;
    if (i < n) out[i] = (bf16)in[i];
}

// in [K,N] f32 row-major -> out [N,K] bf16 row-major
__global__ void k_cast_t(const float* __restrict__ in, bf16* __restrict__ out, int K, int N)
{
    long i = (long)blockIdx.x * blockDim.x + threadIdx.x;
    if (i < (long)K * N) {
        int k = (int)(i / N), n = (int)(i % N);
        out[(long)n * K + k] = (bf16)in[i];
    }
}

__global__ void k_relucast(const float* __restrict__ in, bf16* __restrict__ out, long n)
{
    long i = (long)blockIdx.x * blockDim.x + threadIdx.x;
    if (i < n) { float v = in[i]; out[i] = (bf16)(v > 0.f ? v : 0.f); }
}

__global__ void k_zero_f32(float* __restrict__ p, long n)
{
    long i = (long)blockIdx.x * blockDim.x + threadIdx.x;
    if (i < n) p[i] = 0.f;
}

__global__ void k_zero_bf16(bf16* __restrict__ p, long n)
{
    long i = (long)blockIdx.x * blockDim.x + threadIdx.x;
    if (i < n) p[i] = (bf16)0.f;
}

// =====================================================================
// bf16 WMMA GEMM:  C[M,N] = A[M,K] * B[N,K]^T  (+bias(N)) (+accum C) (+relu)
// block tile 64x128, 4 waves (2x2), wave tile 32x64 (2x4 WMMA fragments).
// A/B tiles DMA'd into double-buffered LDS by the Tensor Data Mover
// (wave 0 issues, TENSORcnt + workgroup barrier synchronize; TDM zero-fills
// out-of-range rows so no guard branches in the hot loop).
// =====================================================================
__global__ void __launch_bounds__(128)
k_gemm_bf16(const bf16* __restrict__ A, int lda,
            const bf16* __restrict__ B, int ldb,
            const float* __restrict__ bias,
            float* __restrict__ Cf,
            bf16*  __restrict__ Cb,
            bf16*  __restrict__ Ct,   // transposed bf16 store [N,M]
            int M, int N, int K, int act, int accum)
{
    __shared__ bf16 As[2][64 * 32];
    __shared__ bf16 Bs[2][128 * 32];

    const int tid  = threadIdx.x;
    const int wave = tid >> 5;
    const int lane = tid & 31;
    const int hlf  = lane >> 4;     // 0 or 1
    const int l16  = lane & 15;
    const int bm = blockIdx.y * 64;
    const int bn = blockIdx.x * 128;
    const int wm = (wave >> 1) * 32;   // 0 / 32
    const int wn = (wave & 1) * 64;    // 0 / 64

    const unsigned ldsA0 = (unsigned)(unsigned long long)(void*)&As[0][0];
    const unsigned ldsA1 = (unsigned)(unsigned long long)(void*)&As[1][0];
    const unsigned ldsB0 = (unsigned)(unsigned long long)(void*)&Bs[0][0];
    const unsigned ldsB1 = (unsigned)(unsigned long long)(void*)&Bs[1][0];

    v8f acc[2][4];
#pragma unroll
    for (int i = 0; i < 2; i++)
#pragma unroll
        for (int j = 0; j < 4; j++)
#pragma unroll
            for (int e = 0; e < 8; e++) acc[i][j][e] = 0.f;

    const unsigned rowsA = (unsigned)(M - bm);   // grid covers M -> positive
    const unsigned rowsB = (unsigned)(N - bn);

    // prologue: DMA chunk 0 into buffer 0
    if (wave == 0) {
        tdm_load_tile(ldsA0, A + (long)bm * lda, rowsA, (unsigned)K, (unsigned)lda, 64u);
        tdm_load_tile(ldsB0, B + (long)bn * ldb, rowsB, (unsigned)K, (unsigned)ldb, 128u);
    }

    int buf = 0;
    for (int k0 = 0; k0 < K; k0 += 32) {
        if (wave == 0) {
            __builtin_amdgcn_s_wait_tensorcnt(0);   // chunk k0 landed in LDS
        }
        __syncthreads();   // all waves: chunk k0 visible; prev chunk's reads done
        if (wave == 0 && (k0 + 32) < K) {
            // overlap: DMA chunk k0+32 into the other buffer while we compute
            tdm_load_tile(buf ? ldsA0 : ldsA1, A + (long)bm * lda + (k0 + 32), rowsA,
                          (unsigned)K, (unsigned)lda, 64u);
            tdm_load_tile(buf ? ldsB0 : ldsB1, B + (long)bn * ldb + (k0 + 32), rowsB,
                          (unsigned)K, (unsigned)ldb, 128u);
            __builtin_prefetch(A + (long)bm * lda + (k0 + 32), 0, 1);
        }

#pragma unroll
        for (int i = 0; i < 2; i++) {
            // A fragment: lanes 0-15 hold K{0..7,16..23}, lanes 16-31 K{8..15,24..31}
            const bf16* ap = &As[buf][(wm + i * 16 + l16) * 32 + hlf * 8];
            union { v16bf v; v8bf q[2]; } ua;
            ua.q[0] = *(const v8bf*)ap;
            ua.q[1] = *(const v8bf*)(ap + 16);
#pragma unroll
            for (int j = 0; j < 4; j++) {
                const bf16* bp = &Bs[buf][(wn + j * 16 + l16) * 32 + hlf * 8];
                union { v16bf v; v8bf q[2]; } ub;
                ub.q[0] = *(const v8bf*)bp;
                ub.q[1] = *(const v8bf*)(bp + 16);
                acc[i][j] = __builtin_amdgcn_wmma_f32_16x16x32_bf16(
                    false, ua.v, false, ub.v, (short)0, acc[i][j], false, false);
            }
        }
        buf ^= 1;
    }

    // epilogue: C layout — VGPR e, lanes 0-15 row e, lanes 16-31 row e+8
#pragma unroll
    for (int i = 0; i < 2; i++)
#pragma unroll
        for (int j = 0; j < 4; j++)
#pragma unroll
            for (int e = 0; e < 8; e++) {
                int row = bm + wm + i * 16 + e + hlf * 8;
                int col = bn + wn + j * 16 + l16;
                if (row < M && col < N) {
                    float v = acc[i][j][e];
                    if (bias)  v += bias[col];
                    if (accum) v += Cf[(long)row * N + col];
                    if (act)   v = v > 0.f ? v : 0.f;
                    if (Cf) Cf[(long)row * N + col] = v;
                    if (Cb) Cb[(long)row * N + col] = (bf16)v;
                    if (Ct) Ct[(long)col * M + row] = (bf16)v;
                }
            }
}

// =====================================================================
// LSTM cell elementwise (gates = [N,4H] f32), h2 -> bf16 features
// =====================================================================
__global__ void k_lstm(const float* __restrict__ gates,
                       const float* __restrict__ cprev,
                       bf16* __restrict__ hout, int N, int H, int ldo, int coff)
{
    long i = (long)blockIdx.x * blockDim.x + threadIdx.x;
    if (i >= (long)N * H) return;
    int n = (int)(i / H), j = (int)(i % H);
    const float* g = gates + (long)n * 4 * H;
    float ig = 1.f / (1.f + __expf(-g[j]));
    float fg = 1.f / (1.f + __expf(-g[H + j]));
    float gg = tanhf(g[2 * H + j]);
    float og = 1.f / (1.f + __expf(-g[3 * H + j]));
    float c2 = fg * cprev[i] + ig * gg;
    float h2 = og * tanhf(c2);
    hout[(long)n * ldo + coff + j] = (bf16)h2;
}

// =====================================================================
// GAT attention helpers
// =====================================================================
__global__ void k_es(const float* __restrict__ hs, const float* __restrict__ a,
                     float* __restrict__ es, int Ns, int D)
{
    long i = (long)blockIdx.x * blockDim.x + threadIdx.x;
    if (i >= (long)Ns * 4) return;
    int n = (int)(i / 4), h = (int)(i % 4);
    const float* hp = hs + (long)n * 4 * D + h * D;
    const float* ap = a + h * D;
    float s = 0.f;
    for (int d = 0; d < D; d++) s += hp[d] * ap[d];
    es[i] = s;
}

__global__ void k_colmax(const float* __restrict__ es, float* __restrict__ out, int Ns)
{
    __shared__ float sm[256];
    int h = blockIdx.x;
    float m = -3.4e38f;
    for (int n = threadIdx.x; n < Ns; n += 256) m = fmaxf(m, es[n * 4 + h]);
    sm[threadIdx.x] = m;
    __syncthreads();
    for (int s = 128; s > 0; s >>= 1) {
        if (threadIdx.x < s) sm[threadIdx.x] = fmaxf(sm[threadIdx.x], sm[threadIdx.x + s]);
        __syncthreads();
    }
    if (threadIdx.x == 0) out[h] = sm[0];
}

// wvec[k,h] = dot(Wd[k, h*D : h*D+D], a_d[h])   — pre-contract Wd with a_d
__global__ void k_wvec(const float* __restrict__ Wd, const float* __restrict__ ad,
                       float* __restrict__ wv, int K, int D)
{
    long i = (long)blockIdx.x * blockDim.x + threadIdx.x;
    if (i >= (long)K * 4) return;
    int k = (int)(i / 4), h = (int)(i % 4);
    const float* wp = Wd + (long)k * 4 * D + h * D;
    const float* ap = ad + h * D;
    float s = 0.f;
    for (int d = 0; d < D; d++) s += wp[d] * ap[d];
    wv[i] = s;
}

__global__ void k_ed(const bf16* __restrict__ f, int ldf, const float* __restrict__ wv,
                     float* __restrict__ ed, int Nd, int K)
{
    long i = (long)blockIdx.x * blockDim.x + threadIdx.x;
    if (i >= (long)Nd * 4) return;
    int m = (int)(i / 4), h = (int)(i % 4);
    const bf16* fp = f + (long)m * ldf;
    float s = 0.f;
    for (int k = 0; k < K; k++) s += (float)fp[k] * wv[k * 4 + h];
    ed[i] = s;
}

// attention aggregation via WMMA: agg[m, h*D+d] += softmax_n(lrelu(ed+es)) . hs[n, h*D+d]
// one wave per (16-row m-tile, head). hst is bf16 [4*D, Ns] (channel-major).
template <int D>
__global__ void __launch_bounds__(32)
k_attn(const bf16* __restrict__ hst,
       const float* __restrict__ es, const float* __restrict__ ed,
       const float* __restrict__ esmax,
       float* __restrict__ agg,
       int Nd, int Ns, int HD)
{
    const int lane = threadIdx.x;
    const int hlf = lane >> 4, l16 = lane & 15;
    const int h = blockIdx.y;
    const int mbase = blockIdx.x * 16;
    const int m = mbase + l16;
    constexpr int NT = D / 16;

    float edv = (m < Nd) ? ed[m * 4 + h] : 0.f;
    float t = edv + esmax[h];
    float rowmax = t > 0.f ? t : NEG_SLOPE * t;   // lrelu monotone -> exact row max

    v8f acc[NT];
#pragma unroll
    for (int j = 0; j < NT; j++)
#pragma unroll
        for (int e = 0; e < 8; e++) acc[j][e] = 0.f;
    float dsum = 0.f;
    __shared__ float denom[16];

    for (int n0 = 0; n0 < Ns; n0 += 32) {
        v16bf wfrag;
#pragma unroll
        for (int e = 0; e < 16; e++) {
            int kk = (e < 8) ? (hlf * 8 + e) : (8 + hlf * 8 + e);
            int n = n0 + kk;
            float w = 0.f;
            if (m < Nd && n < Ns) {
                float z = edv + es[n * 4 + h];
                z = z > 0.f ? z : NEG_SLOPE * z;
                w = __expf(z - rowmax);
            }
            dsum += w;
            wfrag[e] = (bf16)w;
        }
        const bool full = (n0 + 32 <= Ns);
#pragma unroll
        for (int j = 0; j < NT; j++) {
            int ch = h * D + j * 16 + l16;
            const bf16* bp = hst + (long)ch * Ns + n0;
            union { v16bf v; v8bf q[2]; } ub;
            if (full) {
                ub.q[0] = *(const v8bf*)(bp + hlf * 8);
                ub.q[1] = *(const v8bf*)(bp + 16 + hlf * 8);
            } else {
#pragma unroll
                for (int e = 0; e < 16; e++) {
                    int kk = (e < 8) ? (hlf * 8 + e) : (8 + hlf * 8 + e);
                    ub.v[e] = (n0 + kk < Ns) ? bp[kk] : (bf16)0.f;
                }
            }
            acc[j] = __builtin_amdgcn_wmma_f32_16x16x32_bf16(
                false, wfrag, false, ub.v, (short)0, acc[j], false, false);
        }
    }

    dsum += __shfl_xor(dsum, 16, 32);
    if (hlf == 0) denom[l16] = dsum;
    __syncthreads();

#pragma unroll
    for (int j = 0; j < NT; j++)
#pragma unroll
        for (int e = 0; e < 8; e++) {
            int row = mbase + e + hlf * 8;
            if (row < Nd) {
                float dn = denom[e + hlf * 8];
                float v = acc[j][e] / (dn > 1e-20f ? dn : 1e-20f);
                atomicAdd(&agg[(long)row * HD + h * D + j * 16 + l16], v);
            }
        }
}

// =====================================================================
// output epilogues
// =====================================================================
__global__ void k_outmean(const float* __restrict__ agg, float* __restrict__ out,
                          int N, int D, int H)
{
    long i = (long)blockIdx.x * blockDim.x + threadIdx.x;
    if (i >= (long)N * D) return;
    int n = (int)(i / D), d = (int)(i % D);
    float s = 0.f;
    for (int h = 0; h < H; h++) s += agg[(long)n * H * D + h * D + d];
    out[i] = s * (1.f / H);
}

__global__ void k_critic(const float* __restrict__ aggS,
                         const float* __restrict__ cw, const float* __restrict__ cb,
                         float* __restrict__ out)
{
    if (threadIdx.x == 0 && blockIdx.x == 0) {
        float v = 0.f;
        for (int d = 0; d < 32; d++) {
            float s = 0.f;
            for (int h = 0; h < 4; h++) s += aggS[h * 32 + d];
            v += 0.25f * s * cw[d];
        }
        out[0] = v + cb[0];
    }
}

// =====================================================================
// host-side GAT layer driver
// relations sorted alphabetically (jax dict flatten):
//   A->A, A->P, A->state, P->A, P->P, P->state, state->A, state->P, state->state
// per relation leaf order (sorted keys): Wd, Ws, a_d, a_s
// type ids: 0=P, 1=A, 2=state
// =====================================================================
template <int D>
static void run_gat(const bf16* fP, const bf16* fA, const bf16* fS,
                    const int* ld, const int* din, const int* Nn,
                    void* const* d_in, int base,
                    float* aggP, float* aggA, float* aggS,
                    bf16* wbuf, float* hsf, bf16* hst,
                    float* es, float* ed, float* esmax, float* wvec,
                    hipStream_t stream)
{
    const bf16* feats[3] = { fP, fA, fS };
    float* agg[3] = { aggP, aggA, aggS };
    static const int RS[9] = { 1, 1, 1, 0, 0, 0, 2, 2, 2 };
    static const int RD[9] = { 1, 0, 2, 1, 0, 2, 1, 0, 2 };
    const int HD = 4 * D;

    for (int r = 0; r < 9; r++) {
        int s = RS[r], d = RD[r];
        const float* Wd = (const float*)d_in[base + r * 4 + 0];
        const float* Ws = (const float*)d_in[base + r * 4 + 1];
        const float* ad = (const float*)d_in[base + r * 4 + 2];
        const float* as = (const float*)d_in[base + r * 4 + 3];
        int Ks = din[s], Kd = din[d], Ns = Nn[s], Nd = Nn[d];

        { long n = (long)Ks * HD;
          k_cast_t<<<dim3((unsigned)((n + 255) / 256)), 256, 0, stream>>>(Ws, wbuf, Ks, HD); }
        { dim3 g((HD + 127) / 128, (Ns + 63) / 64);
          k_gemm_bf16<<<g, 128, 0, stream>>>(feats[s], ld[s], wbuf, Ks, nullptr,
                                             hsf, nullptr, hst, Ns, HD, Ks, 0, 0); }
        { long n = (long)Ns * 4;
          k_es<<<dim3((unsigned)((n + 255) / 256)), 256, 0, stream>>>(hsf, as, es, Ns, D); }
        k_colmax<<<4, 256, 0, stream>>>(es, esmax, Ns);
        { long n = (long)Kd * 4;
          k_wvec<<<dim3((unsigned)((n + 255) / 256)), 256, 0, stream>>>(Wd, ad, wvec, Kd, D); }
        { long n = (long)Nd * 4;
          k_ed<<<dim3((unsigned)((n + 255) / 256)), 256, 0, stream>>>(feats[d], ld[d], wvec, ed, Nd, Kd); }
        { dim3 g((Nd + 15) / 16, 4);
          k_attn<D><<<g, 32, 0, stream>>>(hst, es, ed, esmax, agg[d], Nd, Ns, HD); }
    }
}

// =====================================================================
// entry point
// =====================================================================
extern "C" void kernel_launch(void* const* d_in, const int* in_sizes, int n_in,
                              void* d_out, int out_size, void* d_ws, size_t ws_size,
                              hipStream_t stream)
{
    (void)in_sizes; (void)n_in; (void)out_size; (void)ws_size;

    const float* x      = (const float*)d_in[0];
    const float* h_ps   = (const float*)d_in[1];
    const float* c_ps   = (const float*)d_in[2];
    const float* h_po   = (const float*)d_in[3];
    const float* c_po   = (const float*)d_in[4];
    const float* w_stat = (const float*)d_in[5];
    const float* b_stat = (const float*)d_in[6];
    const float* w_obs  = (const float*)d_in[7];
    const float* b_obs  = (const float*)d_in[8];
    const float* s_wih  = (const float*)d_in[9];
    const float* s_whh  = (const float*)d_in[10];
    const float* s_bih  = (const float*)d_in[11];
    const float* s_bhh  = (const float*)d_in[12];
    const float* o_wih  = (const float*)d_in[13];
    const float* o_whh  = (const float*)d_in[14];
    const float* o_bih  = (const float*)d_in[15];
    const float* o_bhh  = (const float*)d_in[16];
    const int GAT1 = 17;            // 36 leaves
    const int GAT2 = 17 + 36;       // 36 leaves
    const float* critic_w = (const float*)d_in[GAT2 + 36 + 0];
    const float* critic_b = (const float*)d_in[GAT2 + 36 + 1];
    float* out = (float*)d_out;

    // ---------------- workspace carving ----------------
    size_t off = 0;
    auto alloc = [&](size_t bytes) -> char* {
        char* p = (char*)d_ws + off;
        off += (bytes + 255) & ~size_t(255);
        return p;
    };
    bf16*  XPS   = (bf16*) alloc((size_t)C_NUM_P * C_DSTAT * 2);
    bf16*  XPO   = (bf16*) alloc((size_t)C_NUM_P * C_DOBS * 2);
    bf16*  XAS   = (bf16*) alloc((size_t)C_NUM_A * C_DOBS * 2);
    bf16*  WBUF  = (bf16*) alloc((size_t)C_DSTAT * C_DSTAT * 2);  // max weight, reused
    bf16*  SBF   = (bf16*) alloc((size_t)C_NUM_P * C_DSTAT * 2);  // relu(prepro_stat)
    bf16*  OBF   = (bf16*) alloc((size_t)C_NUM_P * C_DOBS * 2);   // relu(prepro_obs)
    float* GATS  = (float*)alloc((size_t)C_NUM_P * 4 * C_P_S * 4);
    float* GATO  = (float*)alloc((size_t)C_NUM_P * 4 * C_STATE * 4);
    bf16*  HBUF  = (bf16*) alloc((size_t)C_NUM_P * C_P_S * 2);    // h state bf16, reused
    bf16*  FEATP = (bf16*) alloc((size_t)C_NUM_P * C_IN_P * 2);   // [2048,384]
    bf16*  FEATS = (bf16*) alloc((size_t)C_STATE * 2);            // state feats (zeros)
    float* HSF   = (float*)alloc((size_t)C_NUM_P * 256 * 4);
    bf16*  HST   = (bf16*) alloc((size_t)256 * C_NUM_P * 2);
    float* ES    = (float*)alloc((size_t)C_NUM_P * 4 * 4);
    float* ED    = (float*)alloc((size_t)C_NUM_P * 4 * 4);
    float* ESMAX = (float*)alloc(4 * 4);
    float* WVEC  = (float*)alloc((size_t)C_DOBS * 4 * 4);
    float* AGG1P = (float*)alloc((size_t)C_NUM_P * 256 * 4);
    float* AGG1A = (float*)alloc((size_t)C_NUM_A * 256 * 4);
    float* AGG1S = (float*)alloc((size_t)256 * 4);
    bf16*  H1P   = (bf16*) alloc((size_t)C_NUM_P * 256 * 2);
    bf16*  H1A   = (bf16*) alloc((size_t)C_NUM_A * 256 * 2);
    bf16*  H1S   = (bf16*) alloc((size_t)256 * 2);
    float* AGG2P = (float*)alloc((size_t)C_NUM_P * 128 * 4);
    float* AGG2A = (float*)alloc((size_t)C_NUM_A * 128 * 4);
    float* AGG2S = (float*)alloc((size_t)128 * 4);

    auto blocks = [](long n) { return dim3((unsigned)((n + 255) / 256)); };

    // ---------------- stage 0: gather + zero ----------------
    k_gather<<<2048, 256, 0, stream>>>(x, XPS, XPO, XAS);
    k_zero_bf16<<<1, 256, 0, stream>>>(FEATS, C_STATE);

    // ---------------- stage 1: prepro_stat (relu) ----------------
    k_cast<<<blocks((long)C_DSTAT * C_DSTAT), 256, 0, stream>>>(w_stat, WBUF, (long)C_DSTAT * C_DSTAT);
    { dim3 g(C_DSTAT / 128, C_NUM_P / 64);
      k_gemm_bf16<<<g, 128, 0, stream>>>(XPS, C_DSTAT, WBUF, C_DSTAT, b_stat,
                                         nullptr, SBF, nullptr,
                                         C_NUM_P, C_DSTAT, C_DSTAT, 1, 0); }
    // ---------------- stage 2: lstm_s ----------------
    k_cast<<<blocks((long)4 * C_P_S * C_DSTAT), 256, 0, stream>>>(s_wih, WBUF, (long)4 * C_P_S * C_DSTAT);
    { dim3 g((4 * C_P_S) / 128, C_NUM_P / 64);
      k_gemm_bf16<<<g, 128, 0, stream>>>(SBF, C_DSTAT, WBUF, C_DSTAT, s_bih,
                                         GATS, nullptr, nullptr,
                                         C_NUM_P, 4 * C_P_S, C_DSTAT, 0, 0); }
    k_cast<<<blocks((long)C_NUM_P * C_P_S), 256, 0, stream>>>(h_ps, HBUF, (long)C_NUM_P * C_P_S);
    k_cast<<<blocks((long)4 * C_P_S * C_P_S), 256, 0, stream>>>(s_whh, WBUF, (long)4 * C_P_S * C_P_S);
    { dim3 g((4 * C_P_S) / 128, C_NUM_P / 64);
      k_gemm_bf16<<<g, 128, 0, stream>>>(HBUF, C_P_S, WBUF, C_P_S, s_bhh,
                                         GATS, nullptr, nullptr,
                                         C_NUM_P, 4 * C_P_S, C_P_S, 0, 1); }
    k_lstm<<<blocks((long)C_NUM_P * C_P_S), 256, 0, stream>>>(GATS, c_ps, FEATP, C_NUM_P, C_P_S, C_IN_P, 0);

    // ---------------- stage 3: prepro_obs + lstm_o ----------------
    k_cast<<<blocks((long)C_DOBS * C_DOBS), 256, 0, stream>>>(w_obs, WBUF, (long)C_DOBS * C_DOBS);
    { dim3 g(C_DOBS / 128, C_NUM_P / 64);
      k_gemm_bf16<<<g, 128, 0, stream>>>(XPO, C_DOBS, WBUF, C_DOBS, b_obs,
                                         nullptr, OBF, nullptr,
                                         C_NUM_P, C_DOBS, C_DOBS, 1, 0); }
    k_cast<<<blocks((long)4 * C_STATE * C_DOBS), 256, 0, stream>>>(o_wih, WBUF, (long)4 * C_STATE * C_DOBS);
    { dim3 g((4 * C_STATE) / 128, C_NUM_P / 64);
      k_gemm_bf16<<<g, 128, 0, stream>>>(OBF, C_DOBS, WBUF, C_DOBS, o_bih,
                                         GATO, nullptr, nullptr,
                                         C_NUM_P, 4 * C_STATE, C_DOBS, 0, 0); }
    k_cast<<<blocks((long)C_NUM_P * C_STATE), 256, 0, stream>>>(h_po, HBUF, (long)C_NUM_P * C_STATE);
    k_cast<<<blocks((long)4 * C_STATE * C_STATE), 256, 0, stream>>>(o_whh, WBUF, (long)4 * C_STATE * C_STATE);
    { dim3 g((4 * C_STATE) / 128, C_NUM_P / 64);
      k_gemm_bf16<<<g, 128, 0, stream>>>(HBUF, C_STATE, WBUF, C_STATE, o_bhh,
                                         GATO, nullptr, nullptr,
                                         C_NUM_P, 4 * C_STATE, C_STATE, 0, 1); }
    k_lstm<<<blocks((long)C_NUM_P * C_STATE), 256, 0, stream>>>(GATO, c_po, FEATP, C_NUM_P, C_STATE, C_IN_P, C_P_S);

    // ---------------- stage 4: GAT layer 1 (H=4, D=64, merge=cat) ----------------
    k_zero_f32<<<blocks((long)C_NUM_P * 256), 256, 0, stream>>>(AGG1P, (long)C_NUM_P * 256);
    k_zero_f32<<<blocks((long)C_NUM_A * 256), 256, 0, stream>>>(AGG1A, (long)C_NUM_A * 256);
    k_zero_f32<<<1, 256, 0, stream>>>(AGG1S, 256);
    {
        const int ld1[3]  = { C_IN_P, C_DOBS, C_STATE };
        const int din1[3] = { C_IN_P, C_DOBS, C_STATE };
        const int Nn[3]   = { C_NUM_P, C_NUM_A, 1 };
        run_gat<64>(FEATP, XAS, FEATS, ld1, din1, Nn, d_in, GAT1,
                    AGG1P, AGG1A, AGG1S, WBUF, HSF, HST, ES, ED, ESMAX, WVEC, stream);
    }
    k_relucast<<<blocks((long)C_NUM_P * 256), 256, 0, stream>>>(AGG1P, H1P, (long)C_NUM_P * 256);
    k_relucast<<<blocks((long)C_NUM_A * 256), 256, 0, stream>>>(AGG1A, H1A, (long)C_NUM_A * 256);
    k_relucast<<<1, 256, 0, stream>>>(AGG1S, H1S, 256);

    // ---------------- stage 5: GAT layer 2 (H=4, D=32, merge=avg) ----------------
    k_zero_f32<<<blocks((long)C_NUM_P * 128), 256, 0, stream>>>(AGG2P, (long)C_NUM_P * 128);
    k_zero_f32<<<blocks((long)C_NUM_A * 128), 256, 0, stream>>>(AGG2A, (long)C_NUM_A * 128);
    k_zero_f32<<<1, 128, 0, stream>>>(AGG2S, 128);
    {
        const int ld2[3]  = { 256, 256, 256 };
        const int din2[3] = { 256, 256, 256 };
        const int Nn[3]   = { C_NUM_P, C_NUM_A, 1 };
        run_gat<32>(H1P, H1A, H1S, ld2, din2, Nn, d_in, GAT2,
                    AGG2P, AGG2A, AGG2S, WBUF, HSF, HST, ES, ED, ESMAX, WVEC, stream);
    }

    // ---------------- stage 6: outputs ----------------
    k_outmean<<<blocks((long)C_NUM_P * 32), 256, 0, stream>>>(AGG2P, out, C_NUM_P, 32, 4);
    k_outmean<<<blocks((long)C_NUM_A * 32), 256, 0, stream>>>(AGG2A, out + (long)C_NUM_P * 32, C_NUM_A, 32, 4);
    k_critic<<<1, 32, 0, stream>>>(AGG2S, critic_w, critic_b, out + (long)C_NUM_P * 32 + (long)C_NUM_A * 32);
}
